// MultiHeadAttentionBase_6803228197596
// MI455X (gfx1250) — matrix operand order repaired and checked by reading, not compiled, not run on hardware
//
#include <hip/hip_runtime.h>

// Multi-head attention (B=4, S=2048, D=1024, H=16, Dh=64), flash-attention style.
// bf16 WMMA (v_wmma_f32_16x16x32_bf16), fp32 accumulation, online softmax in
// log2 domain, row-sums via WMMA against a ones matrix, async-DMA K+V tiles.

#define BB   4
#define SS   2048
#define DD   1024
#define HH   16
#define DHH  64
// (1/sqrt(64)) * log2(e): folded into Q during bf16 conversion so softmax uses native exp2.
#define QSCALE 0.18033688011112042f

typedef __attribute__((ext_vector_type(16))) __bf16        v16bf;
typedef __attribute__((ext_vector_type(8)))  float         v8f;
typedef __attribute__((ext_vector_type(4)))  unsigned int  v4u;
typedef __attribute__((ext_vector_type(4)))  float         v4f;
typedef __attribute__((ext_vector_type(4)))  unsigned short v4us;

union FragBF {
    v16bf bf;
    v4u   u4[2];
    unsigned short us[16];
};

__device__ __forceinline__ unsigned short f2bf(float f) {
    unsigned u = __float_as_uint(f);
    u += 0x7FFFu + ((u >> 16) & 1u);   // round-to-nearest-even
    return (unsigned short)(u >> 16);
}

// Pack two f32 -> two bf16 in one u32 (low = a, high = b).
#if __has_builtin(__builtin_amdgcn_cvt_pk_bf16_f32)
__device__ __forceinline__ unsigned pk_bf16(float a, float b) {
    auto p = __builtin_amdgcn_cvt_pk_bf16_f32(a, b);
    return __builtin_bit_cast(unsigned, p);
}
#else
__device__ __forceinline__ unsigned pk_bf16(float a, float b) {
    return (unsigned)f2bf(a) | ((unsigned)f2bf(b) << 16);
}
#endif

#if __has_builtin(__builtin_amdgcn_exp2f)
#define EXP2F(x) __builtin_amdgcn_exp2f(x)
#else
#define EXP2F(x) exp2f(x)
#endif

// ---------------------------------------------------------------------------
// Cross-lane xor-max within 16-lane rows, VALU-only (v_permlane16_b32).
// Selector nibbles: dst lane i reads src lane sel[i]; sel[i] = i ^ mask.
// ---------------------------------------------------------------------------
#if __has_builtin(__builtin_amdgcn_permlane16)
__device__ __forceinline__ float xshuf16(float x, unsigned lo, unsigned hi) {
    int y = __builtin_amdgcn_permlane16(__float_as_int(x), __float_as_int(x),
                                        lo, hi, false, false);
    return __int_as_float(y);
}
__device__ __forceinline__ float rowmax16(float x) {
    x = fmaxf(x, xshuf16(x, 0x67452301u, 0xEFCDAB89u));  // xor 1
    x = fmaxf(x, xshuf16(x, 0x54761032u, 0xDCFE98BAu));  // xor 2
    x = fmaxf(x, xshuf16(x, 0x32107654u, 0xBA98FEDCu));  // xor 4
    x = fmaxf(x, xshuf16(x, 0xFEDCBA98u, 0x76543210u));  // xor 8
    return x;
}
#else
__device__ __forceinline__ float rowmax16(float x) {
    for (int off = 1; off < 16; off <<= 1)
        x = fmaxf(x, __shfl_xor(x, off, 32));
    return x;
}
#endif

// ---------------------------------------------------------------------------
// Pass 1: fp32 -> bf16 conversion with optional scale (bandwidth pass).
// ---------------------------------------------------------------------------
__global__ __launch_bounds__(256) void cvt_f32_bf16(const float* __restrict__ src,
                                                    unsigned short* __restrict__ dst,
                                                    int n4, float scale) {
    int i = blockIdx.x * blockDim.x + threadIdx.x;
    if (i < n4) {
        v4f f = ((const v4f*)src)[i];
        v4us o;
        o.x = f2bf(f.x * scale); o.y = f2bf(f.y * scale);
        o.z = f2bf(f.z * scale); o.w = f2bf(f.w * scale);
        ((v4us*)dst)[i] = o;
    }
}

// 64B async global->LDS copy (4 x b128; IOFFSET applies to both addresses).
__device__ __forceinline__ void async_copy64B(unsigned lds, const unsigned short* g) {
    unsigned long long ga = (unsigned long long)(size_t)g;
    asm volatile("global_load_async_to_lds_b128 %0, %1, off"           :: "v"(lds), "v"(ga) : "memory");
    asm volatile("global_load_async_to_lds_b128 %0, %1, off offset:16" :: "v"(lds), "v"(ga) : "memory");
    asm volatile("global_load_async_to_lds_b128 %0, %1, off offset:32" :: "v"(lds), "v"(ga) : "memory");
    asm volatile("global_load_async_to_lds_b128 %0, %1, off offset:48" :: "v"(lds), "v"(ga) : "memory");
}

// ---------------------------------------------------------------------------
// Pass 2: flash attention. Block = 128 threads = 4 waves; wave owns 16 query
// rows; KV tiles of 64 keys; double-buffered async K+V tiles; 18 WMMAs/iter.
// ---------------------------------------------------------------------------
__global__ __launch_bounds__(128) void fa_kernel(const unsigned short* __restrict__ qbf,
                                                 const unsigned short* __restrict__ kbf,
                                                 const unsigned short* __restrict__ vbf,
                                                 float* __restrict__ out) {
    __shared__ unsigned short kTile[2][64 * DHH];   // double-buffered K [key][d], 2 x 8 KB
    __shared__ unsigned short vTile[2][64 * DHH];   // double-buffered V [key][d], 2 x 8 KB
    __shared__ unsigned short pTile[4][16 * 72];    // per-wave P staging, 144B rows (16B-aligned)

    const int bh   = blockIdx.y;
    const int b    = bh / HH;
    const int h    = bh % HH;
    const int q0   = blockIdx.x * 64;
    const int tid  = threadIdx.x;
    const int wave = tid >> 5;
    const int lane = tid & 31;
    const int lhalf = lane >> 4;
    const int l16   = lane & 15;
    const int mrow  = q0 + wave * 16;
    // P staging interleaves columns (staged 2j <- n=j, 2j+1 <- n=j+16); the PV
    // B-operand compensates by permuting V rows with the same permutation.
    const int vperm = (lane >> 1) + ((lane & 1) << 4);

    const unsigned short* qbase = qbf + (size_t)b * SS * DD + (size_t)h * DHH;
    const unsigned short* kbase = kbf + (size_t)b * SS * DD + (size_t)h * DHH;
    const unsigned short* vbase = vbf + (size_t)b * SS * DD + (size_t)h * DHH;

    // Q A-fragments (16x32 each). A layout: lane -> M=l16; halves 0..7 -> K=lhalf*8+{0..7},
    // halves 8..15 -> K=16+lhalf*8+{0..7}.
    FragBF qa[2];
    {
        const unsigned short* qrow = qbase + (size_t)(mrow + l16) * DD;
        const int koff = lhalf * 8;
#pragma unroll
        for (int c = 0; c < 2; ++c) {
            qa[c].u4[0] = *(const v4u*)(qrow + c * 32 + koff);
            qa[c].u4[1] = *(const v4u*)(qrow + c * 32 + 16 + koff);
        }
    }

    // Ones B-matrix (32x16 of bf16 1.0) for WMMA row sums.
    FragBF onesB;
#pragma unroll
    for (int i = 0; i < 2; ++i)
        onesB.u4[i] = (v4u){0x3F803F80u, 0x3F803F80u, 0x3F803F80u, 0x3F803F80u};

    // Accumulators (C/D layout: lane -> N=l16, VGPR r -> M = r + 8*lhalf).
    v8f acc[4], accS;
#pragma unroll
    for (int t = 0; t < 4; ++t)
        acc[t] = (v8f){0.f, 0.f, 0.f, 0.f, 0.f, 0.f, 0.f, 0.f};
    accS = (v8f){0.f, 0.f, 0.f, 0.f, 0.f, 0.f, 0.f, 0.f};
    float mrun[8];
#pragma unroll
    for (int r = 0; r < 8; ++r) mrun[r] = -1e30f;

    // K/V-tile copy assignment: 128 threads x 64B per tensor.
    const int krow = tid >> 1;
    const int kcol = (tid & 1) * 32;
    const unsigned short* kcopy = kbase + (size_t)krow * DD + kcol;
    const unsigned short* vcopy = vbase + (size_t)krow * DD + kcol;
    const unsigned kldsA = (unsigned)(size_t)(&kTile[0][krow * DHH + kcol]);
    const unsigned kldsB = (unsigned)(size_t)(&kTile[1][krow * DHH + kcol]);
    const unsigned vldsA = (unsigned)(size_t)(&vTile[0][krow * DHH + kcol]);
    const unsigned vldsB = (unsigned)(size_t)(&vTile[1][krow * DHH + kcol]);

    // Prefetch tile 0 (K and V).
    async_copy64B(kldsA, kcopy);
    async_copy64B(vldsA, vcopy);

    const int NIT = SS / 64;
    for (int it = 0; it < NIT; ++it) {
        asm volatile("s_wait_asynccnt 0" ::: "memory");
        __syncthreads();
        const int buf = it & 1;
        if (it + 1 < NIT) {
            const size_t adv = (size_t)(it + 1) * 64 * DD;
            async_copy64B(buf ? kldsA : kldsB, kcopy + adv);
            async_copy64B(buf ? vldsA : vldsB, vcopy + adv);
        }

        // Scores: 4 n-tiles of 16x16, K^T B-fragments via plain ds_load_b128 reads of K rows.
        v8f st[4];
#pragma unroll
        for (int t = 0; t < 4; ++t) {
            FragBF kb0, kb1;
            unsigned a0 = (unsigned)(size_t)(&kTile[buf][(t * 16 + l16) * DHH + lhalf * 8]);
            asm volatile("ds_load_b128 %0, %1"           : "=v"(kb0.u4[0]) : "v"(a0));
            asm volatile("ds_load_b128 %0, %1 offset:32" : "=v"(kb0.u4[1]) : "v"(a0));
            asm volatile("ds_load_b128 %0, %1 offset:64" : "=v"(kb1.u4[0]) : "v"(a0));
            asm volatile("ds_load_b128 %0, %1 offset:96" : "=v"(kb1.u4[1]) : "v"(a0));
            asm volatile("s_wait_dscnt 0" ::: "memory");
            v8f c = (v8f){0.f, 0.f, 0.f, 0.f, 0.f, 0.f, 0.f, 0.f};
            c = __builtin_amdgcn_wmma_f32_16x16x32_bf16(false, qa[0].bf, false, kb0.bf,
                                                        (short)0, c, false, false);
            c = __builtin_amdgcn_wmma_f32_16x16x32_bf16(false, qa[1].bf, false, kb1.bf,
                                                        (short)0, c, false, false);
            st[t] = c;
        }

        // Online softmax (log2 domain; scale pre-folded into Q).
#pragma unroll
        for (int r = 0; r < 8; ++r) {
            float s0 = st[0][r], s1 = st[1][r], s2 = st[2][r], s3 = st[3][r];
            float mx = rowmax16(fmaxf(fmaxf(s0, s1), fmaxf(s2, s3)));
            float mnew  = fmaxf(mrun[r], mx);
            float alpha = EXP2F(mrun[r] - mnew);
            mrun[r] = mnew;
            float p0 = EXP2F(s0 - mnew);
            float p1 = EXP2F(s1 - mnew);
            float p2 = EXP2F(s2 - mnew);
            float p3 = EXP2F(s3 - mnew);
#pragma unroll
            for (int t = 0; t < 4; ++t)
                acc[t][r] *= alpha;
            accS[r] *= alpha;
            // Stage P (packed bf16 pairs) into wave-private LDS, interleaved columns.
            const int m = r + 8 * lhalf;
            unsigned* pw = (unsigned*)(&pTile[wave][m * 72 + 2 * l16]);
            pw[0]  = pk_bf16(p0, p1);   // staged cols 2j, 2j+1   (n = j, j+16)
            pw[16] = pk_bf16(p2, p3);   // staged cols 32+2j, 33+2j (n = j+32, j+48)
        }

        // Re-load P as A-fragments (16x32 each); wave-private LDS, DS in-order per wave.
        FragBF pa[2];
        {
            const int koff = lhalf * 8;
#pragma unroll
            for (int c = 0; c < 2; ++c) {
                const unsigned short* pw = &pTile[wave][l16 * 72 + c * 32 + koff];
                pa[c].u4[0] = *(const v4u*)(pw);
                pa[c].u4[1] = *(const v4u*)(pw + 16);
            }
        }

        // Row sums via WMMA: accS += P x ones.
        accS = __builtin_amdgcn_wmma_f32_16x16x32_bf16(false, pa[0].bf, false, onesB.bf,
                                                       (short)0, accS, false, false);
        accS = __builtin_amdgcn_wmma_f32_16x16x32_bf16(false, pa[1].bf, false, onesB.bf,
                                                       (short)0, accS, false, false);

        // PV: acc[t] += P (16x64) * V (64x16). V B-fragments from the shared LDS tile:
        // lane -> N(d), elements -> K(row)=key (permuted to match staged-P column order).
#pragma unroll
        for (int t = 0; t < 4; ++t) {
#pragma unroll
            for (int c = 0; c < 2; ++c) {
                FragBF vb;
                const unsigned short* vr = &vTile[buf][(c * 32) * DHH + t * 16 + l16];
#pragma unroll
                for (int s = 0; s < 16; ++s) {
                    const int q = (s < 8) ? (lhalf * 8 + s) : (16 + lhalf * 8 + s - 8);
                    vb.us[s] = vr[((q >> 1) + ((q & 1) << 4)) * DHH];
                }
                acc[t] = __builtin_amdgcn_wmma_f32_16x16x32_bf16(false, pa[c].bf, false, vb.bf,
                                                                 (short)0, acc[t], false, false);
            }
        }
    }

    // Normalize and store fp32 output: out[b, s, h*64 + d].
    float inv[8];
#pragma unroll
    for (int r = 0; r < 8; ++r)
        inv[r] = 1.0f / accS[r];
    float* obase = out + (size_t)b * SS * DD + (size_t)h * DHH;
#pragma unroll
    for (int t = 0; t < 4; ++t) {
#pragma unroll
        for (int r = 0; r < 8; ++r) {
            const int m = r + 8 * lhalf;
            obase[(size_t)(mrow + m) * DD + t * 16 + l16] = acc[t][r] * inv[r];
        }
    }
}

// ---------------------------------------------------------------------------
extern "C" void kernel_launch(void* const* d_in, const int* in_sizes, int n_in,
                              void* d_out, int out_size, void* d_ws, size_t ws_size,
                              hipStream_t stream) {
    const float* q = (const float*)d_in[0];
    const float* k = (const float*)d_in[1];
    const float* v = (const float*)d_in[2];
    float* out = (float*)d_out;

    const size_t N = (size_t)BB * SS * DD;
    unsigned short* qbf = (unsigned short*)d_ws;
    unsigned short* kbf = qbf + N;
    unsigned short* vbf = kbf + N;

    const int n4 = (int)(N / 4);
    const int cblocks = (n4 + 255) / 256;
    cvt_f32_bf16<<<cblocks, 256, 0, stream>>>(q, qbf, n4, (float)QSCALE);
    cvt_f32_bf16<<<cblocks, 256, 0, stream>>>(k, kbf, n4, 1.0f);
    cvt_f32_bf16<<<cblocks, 256, 0, stream>>>(v, vbf, n4, 1.0f);

    dim3 grid(SS / 64, BB * HH);
    fa_kernel<<<grid, 128, 0, stream>>>(qbf, kbf, vbf, out);
}
